// GMRF_46823733461544
// MI455X (gfx1250) — matrix operand order repaired and checked by reading, not compiled, not run on hardware
//
#include <hip/hip_runtime.h>
#include <stdint.h>

// ---------- types ----------
typedef __bf16  v16bf __attribute__((ext_vector_type(16)));
typedef float   v8f   __attribute__((ext_vector_type(8)));
typedef unsigned int v4u __attribute__((ext_vector_type(4)));
typedef int          v8i __attribute__((ext_vector_type(8)));
typedef int          v4i __attribute__((ext_vector_type(4)));

union FragAB { uint32_t u[8]; v16bf v; uint4 q[2]; };

// ---------- bf16 helpers (bit-level, RNE) ----------
__device__ __forceinline__ unsigned short f2bf(float f) {
    union { float f; uint32_t u; } a; a.f = f;
    uint32_t u = a.u;
    uint32_t rnd = 0x7FFFu + ((u >> 16) & 1u);
    return (unsigned short)((u + rnd) >> 16);
}
__device__ __forceinline__ float bf2f(unsigned short h) {
    union { uint32_t u; float f; } a; a.u = ((uint32_t)h) << 16;
    return a.f;
}
__device__ __forceinline__ uint32_t pack2(float x, float y) {
    return (uint32_t)f2bf(x) | ((uint32_t)f2bf(y) << 16);
}
__device__ __forceinline__ void atomicAddF(float* p, float v) {
    __hip_atomic_fetch_add(p, v, __ATOMIC_RELAXED, __HIP_MEMORY_SCOPE_AGENT);
}

// ---------- weight packing: f32 [K,64] -> WMMA-B fragment order ----------
// out[((kt*4+nt)*32+lane)*8+v] = bf16 pair (k,k+1) with
//   k = kt*32 + (v>>2)*16 + (lane>>4)*8 + 2(v&3), n = nt*16 + (lane&15)
__global__ void pack_frag_kernel(const float* __restrict__ W, uint32_t* __restrict__ out,
                                 int K, int KT) {
    int i = blockIdx.x * blockDim.x + threadIdx.x;
    if (i >= KT * 1024) return;
    int v = i & 7, lane = (i >> 3) & 31, nt = (i >> 8) & 3, kt = i >> 10;
    int n = nt * 16 + (lane & 15);
    int k = kt * 32 + (v & 4) * 4 + (lane >> 4) * 8 + 2 * (v & 3);
    float a = (k     < K) ? W[k * 64 + n]       : 0.0f;
    float b = (k + 1 < K) ? W[(k + 1) * 64 + n] : 0.0f;
    out[i] = pack2(a, b);
}

// ---------- embedding: h = nf @ Wemb + bemb -> bf16 [N,64] ----------
__global__ void embed_kernel(const float* __restrict__ nf, const float* __restrict__ Wemb,
                             const float* __restrict__ bemb, unsigned short* __restrict__ hOut,
                             int N) {
    int idx = blockIdx.x * blockDim.x + threadIdx.x;
    if (idx >= N * 64) return;
    int n = idx >> 6, j = idx & 63;
    float s = bemb[j];
#pragma unroll
    for (int k = 0; k < 8; k++) s += nf[n * 8 + k] * Wemb[k * 64 + j];
    hOut[idx] = f2bf(s);
}

// ---------- edge MLP + scatter-add; one wave per 16-edge tile ----------
// Weights (28KB, fragment-ordered, W1|W2 contiguous) DMA'd to LDS via TDM.
#define WAVES 4
#define EW_DW (5 * 1024 + 2 * 1024)       /* 7168 dwords = 28KB per layer */
#define ETILE_DW (16 * 80 + 16 * 32)      /* per-wave A tile + mid tile   */

__global__ __launch_bounds__(128)
void edge_mlp_kernel(const unsigned short* __restrict__ hIn,   // [N,64] bf16
                     const float* __restrict__ ea,             // [E,4]
                     const int* __restrict__ esrc, const int* __restrict__ edst,
                     const uint32_t* __restrict__ Wef,         // [7168] frag-ordered W1|W2
                     const float* __restrict__ b1, const float* __restrict__ b2,
                     float* __restrict__ aggr, int N, int E) {
    __shared__ uint32_t shm[EW_DW + WAVES * ETILE_DW];
    uint32_t* ldsW1 = shm;                 // 5120 dwords
    uint32_t* ldsW2 = shm + 5 * 1024;      // 2048 dwords
    const int wave = threadIdx.x >> 5;
    const int lane = threadIdx.x & 31;
    uint32_t* Atile = shm + EW_DW + wave * ETILE_DW;
    uint32_t* Mid   = Atile + 16 * 80;
    const int e0 = (blockIdx.x * WAVES + wave) * 16;

    // ---- stage A tile [16][160] bf16: [h_dst | h_src | edge_attr | pad] ----
    {
        int r = lane >> 1, half = lane & 1;
        int e = e0 + r; if (e >= E) e = E - 1;
        if (half == 0) {
            const uint4* sp = (const uint4*)(hIn + (size_t)edst[e] * 64);
            uint4* dp = (uint4*)(Atile + r * 80);
            dp[0] = sp[0]; dp[1] = sp[1]; dp[2] = sp[2]; dp[3] = sp[3];
        } else {
            const uint4* sp = (const uint4*)(hIn + (size_t)esrc[e] * 64);
            uint4* dp = (uint4*)(Atile + r * 80 + 32);
            dp[0] = sp[0]; dp[1] = sp[1]; dp[2] = sp[2]; dp[3] = sp[3];
            float4 av = ((const float4*)ea)[e];
            Atile[r * 80 + 64] = pack2(av.x, av.y);
            Atile[r * 80 + 65] = pack2(av.z, av.w);
#pragma unroll
            for (int k = 66; k < 80; k++) Atile[r * 80 + k] = 0u;
        }
    }

    // ---- stage weights (28KB) into LDS once per block via Tensor Data Mover ----
#if __has_builtin(__builtin_amdgcn_tensor_load_to_lds) && __has_builtin(__builtin_amdgcn_s_wait_tensorcnt)
    if (wave == 0) {
        // Tensor DMA descriptor: 1-row tensor of EW_DW dwords -> LDS offset 0
        uint64_t ga = (uint64_t)(uintptr_t)Wef;
        v4u g0;
        g0[0] = 1u;                                   // count=1, user mode
        g0[1] = 0u;                                   // lds_addr = 0 (shm base)
        g0[2] = (uint32_t)ga;                         // global_addr[31:0] (bit64+)
        g0[3] = ((uint32_t)(ga >> 32) & 0x01FFFFFFu)  // global_addr[56:32]
                | 0x80000000u;                        // type = 2 ("image")
        v8i g1;
        g1[0] = (int)0x00020000u;                     // data_size=4B; no mcast
        g1[1] = (int)((EW_DW & 0xFFFFu) << 16);       // tensor_dim0[15:0]
        g1[2] = (int)((EW_DW >> 16) | (1u << 16));    // tensor_dim0[31:16] | tensor_dim1[15:0]=1
        g1[3] = (int)((EW_DW & 0xFFFFu) << 16);       // tile_dim0 = EW_DW
        g1[4] = 1;                                    // tile_dim1 = 1, tile_dim2 = 0
        g1[5] = EW_DW;                                // tensor_dim0_stride[31:0]
        g1[6] = 0;
        g1[7] = 0;
        v4i g2 = {0, 0, 0, 0};
        v4i g3 = {0, 0, 0, 0};
        v8i g4 = {0, 0, 0, 0, 0, 0, 0, 0};
        __builtin_amdgcn_tensor_load_to_lds(g0, g1, g2, g3, g4, 0);
        __builtin_amdgcn_s_wait_tensorcnt(0);
    }
#else
    for (int i = threadIdx.x; i < EW_DW / 4; i += 128)
        ((uint4*)shm)[i] = ((const uint4*)Wef)[i];
#endif
    __syncthreads();

    const int m = lane & 15;

    // ---- GEMM1: [16,160] x [160,64], K-tiles of 32, bias in C ----
    v8f acc[4];
#pragma unroll
    for (int nt = 0; nt < 4; nt++) {
        float bb = b1[nt * 16 + m];
#pragma unroll
        for (int r8 = 0; r8 < 8; r8++) acc[nt][r8] = bb;
    }
    const int kh4 = (lane >> 4) * 4;
#pragma unroll
    for (int kt = 0; kt < 5; kt++) {
        FragAB a;
#pragma unroll
        for (int v = 0; v < 4; v++) a.u[v]     = Atile[m * 80 + kt * 16 + kh4 + v];
#pragma unroll
        for (int v = 0; v < 4; v++) a.u[4 + v] = Atile[m * 80 + kt * 16 + 8 + kh4 + v];
#pragma unroll
        for (int nt = 0; nt < 4; nt++) {
            const uint4* bp = (const uint4*)(ldsW1 + (((kt * 4 + nt) * 32 + lane) << 3));
            FragAB b; b.q[0] = bp[0]; b.q[1] = bp[1];
            acc[nt] = __builtin_amdgcn_wmma_f32_16x16x32_bf16(
                false, a.v, false, b.v, (short)0, acc[nt], false, false);
        }
    }

    // ---- ReLU -> Mid tile [16][64] bf16 ----
    {
        unsigned short* midH = (unsigned short*)Mid;
        int mb = (lane >> 4) * 8;
#pragma unroll
        for (int nt = 0; nt < 4; nt++)
#pragma unroll
            for (int r8 = 0; r8 < 8; r8++) {
                float vv = acc[nt][r8]; vv = vv > 0.0f ? vv : 0.0f;
                midH[(mb + r8) * 64 + nt * 16 + m] = f2bf(vv);
            }
    }
    __syncthreads();

    // ---- GEMM2: [16,64] x [64,64] + bias ----
    v8f acc2[4];
#pragma unroll
    for (int nt = 0; nt < 4; nt++) {
        float bb = b2[nt * 16 + m];
#pragma unroll
        for (int r8 = 0; r8 < 8; r8++) acc2[nt][r8] = bb;
    }
#pragma unroll
    for (int kt = 0; kt < 2; kt++) {
        FragAB a;
#pragma unroll
        for (int v = 0; v < 4; v++) a.u[v]     = Mid[m * 32 + kt * 16 + kh4 + v];
#pragma unroll
        for (int v = 0; v < 4; v++) a.u[4 + v] = Mid[m * 32 + kt * 16 + 8 + kh4 + v];
#pragma unroll
        for (int nt = 0; nt < 4; nt++) {
            const uint4* bp = (const uint4*)(ldsW2 + (((kt * 4 + nt) * 32 + lane) << 3));
            FragAB b; b.q[0] = bp[0]; b.q[1] = bp[1];
            acc2[nt] = __builtin_amdgcn_wmma_f32_16x16x32_bf16(
                false, a.v, false, b.v, (short)0, acc2[nt], false, false);
        }
    }

    // ---- scatter-add into aggr[dst] (f32 hardware atomics, L2-resident) ----
    {
        int mb = (lane >> 4) * 8;
#pragma unroll
        for (int r8 = 0; r8 < 8; r8++) {
            int e = e0 + mb + r8;
            if (e < E) {
                float* dstRow = aggr + (size_t)edst[e] * 64;
#pragma unroll
                for (int nt = 0; nt < 4; nt++)
                    atomicAddF(dstRow + nt * 16 + m, acc2[nt][r8]);
            }
        }
    }
}

// ---------- node MLP: h' = MLP(concat(h, aggr)); one wave per 16-node tile ----------
__global__ __launch_bounds__(128)
void node_mlp_kernel(const unsigned short* __restrict__ hIn, const float* __restrict__ aggr,
                     const uint32_t* __restrict__ W1f,  // frag-ordered, KT=4
                     const float* __restrict__ b1,
                     const uint32_t* __restrict__ W2f,  // frag-ordered, KT=2
                     const float* __restrict__ b2,
                     unsigned short* __restrict__ hOut, int N) {
    __shared__ uint32_t shm[WAVES * (16 * 64 + 16 * 32 + 16 * 32)];
    const int wave = threadIdx.x >> 5;
    const int lane = threadIdx.x & 31;
    uint32_t* Atile = shm + wave * (16 * 64 + 16 * 32 + 16 * 32);
    uint32_t* Mid   = Atile + 16 * 64;
    uint32_t* OutT  = Mid + 16 * 32;
    const int n0 = (blockIdx.x * WAVES + wave) * 16;

    // ---- stage A tile [16][128] bf16: [h | bf16(aggr)] ----
    {
        int r = lane >> 1, half = lane & 1;
        int node = n0 + r; if (node >= N) node = N - 1;
        if (half == 0) {
            const uint4* sp = (const uint4*)(hIn + (size_t)node * 64);
            uint4* dp = (uint4*)(Atile + r * 64);
            dp[0] = sp[0]; dp[1] = sp[1]; dp[2] = sp[2]; dp[3] = sp[3];
        } else {
            const float4* ap = (const float4*)(aggr + (size_t)node * 64);
            uint32_t* dp = Atile + r * 64 + 32;
#pragma unroll
            for (int q = 0; q < 16; q++) {
                float4 f = ap[q];
                dp[q * 2]     = pack2(f.x, f.y);
                dp[q * 2 + 1] = pack2(f.z, f.w);
            }
        }
    }
    __syncthreads();

    const int m = lane & 15;
    const int kh4 = (lane >> 4) * 4;

    // ---- GEMM1: [16,128] x [128,64] ----
    v8f acc[4];
#pragma unroll
    for (int nt = 0; nt < 4; nt++) {
        float bb = b1[nt * 16 + m];
#pragma unroll
        for (int r8 = 0; r8 < 8; r8++) acc[nt][r8] = bb;
    }
#pragma unroll
    for (int kt = 0; kt < 4; kt++) {
        FragAB a;
#pragma unroll
        for (int v = 0; v < 4; v++) a.u[v]     = Atile[m * 64 + kt * 16 + kh4 + v];
#pragma unroll
        for (int v = 0; v < 4; v++) a.u[4 + v] = Atile[m * 64 + kt * 16 + 8 + kh4 + v];
#pragma unroll
        for (int nt = 0; nt < 4; nt++) {
            const uint4* bp = (const uint4*)(W1f + (((kt * 4 + nt) * 32 + lane) << 3));
            FragAB b; b.q[0] = bp[0]; b.q[1] = bp[1];
            acc[nt] = __builtin_amdgcn_wmma_f32_16x16x32_bf16(
                false, a.v, false, b.v, (short)0, acc[nt], false, false);
        }
    }

    // ---- ReLU -> Mid ----
    {
        unsigned short* midH = (unsigned short*)Mid;
        int mb = (lane >> 4) * 8;
#pragma unroll
        for (int nt = 0; nt < 4; nt++)
#pragma unroll
            for (int r8 = 0; r8 < 8; r8++) {
                float vv = acc[nt][r8]; vv = vv > 0.0f ? vv : 0.0f;
                midH[(mb + r8) * 64 + nt * 16 + m] = f2bf(vv);
            }
    }
    __syncthreads();

    // ---- GEMM2: [16,64] x [64,64] + bias -> OutT (bf16) ----
    v8f acc2[4];
#pragma unroll
    for (int nt = 0; nt < 4; nt++) {
        float bb = b2[nt * 16 + m];
#pragma unroll
        for (int r8 = 0; r8 < 8; r8++) acc2[nt][r8] = bb;
    }
#pragma unroll
    for (int kt = 0; kt < 2; kt++) {
        FragAB a;
#pragma unroll
        for (int v = 0; v < 4; v++) a.u[v]     = Mid[m * 32 + kt * 16 + kh4 + v];
#pragma unroll
        for (int v = 0; v < 4; v++) a.u[4 + v] = Mid[m * 32 + kt * 16 + 8 + kh4 + v];
#pragma unroll
        for (int nt = 0; nt < 4; nt++) {
            const uint4* bp = (const uint4*)(W2f + (((kt * 4 + nt) * 32 + lane) << 3));
            FragAB b; b.q[0] = bp[0]; b.q[1] = bp[1];
            acc2[nt] = __builtin_amdgcn_wmma_f32_16x16x32_bf16(
                false, a.v, false, b.v, (short)0, acc2[nt], false, false);
        }
    }
    {
        unsigned short* outH = (unsigned short*)OutT;
        int mb = (lane >> 4) * 8;
#pragma unroll
        for (int nt = 0; nt < 4; nt++)
#pragma unroll
            for (int r8 = 0; r8 < 8; r8++)
                outH[(mb + r8) * 64 + nt * 16 + m] = f2bf(acc2[nt][r8]);
    }
    __syncthreads();

    // ---- coalesced store of h' ----
    {
        int r = lane >> 1, half = lane & 1;
        int node = n0 + r;
        if (node < N) {
            const uint4* sp = (const uint4*)(OutT + r * 32 + half * 16);
            uint4* dp = (uint4*)(hOut + (size_t)node * 64 + half * 32);
            dp[0] = sp[0]; dp[1] = sp[1];
        }
    }
}

// ---------- factor values v, sparse matvec Lz, log-det partial ----------
__global__ void factor_kernel(const unsigned short* __restrict__ h,
                              const float* __restrict__ W_L, const float* __restrict__ b_L,
                              const int* __restrict__ Lrow, const int* __restrict__ Lcol,
                              const float* __restrict__ x, const float* __restrict__ mean,
                              float* __restrict__ Lz, float* __restrict__ accum,
                              int N, int M) {
    int i = blockIdx.x * blockDim.x + threadIdx.x;
    float logv = 0.0f;
    if (i < M) {
        int r = Lrow[i], c = Lcol[i];
        float s = b_L[0];
        const uint4* hr = (const uint4*)(h + (size_t)r * 64);
        const uint4* hc = (const uint4*)(h + (size_t)c * 64);
#pragma unroll
        for (int q = 0; q < 8; q++) {
            uint4 w = hr[q];
            uint32_t d[4] = { w.x, w.y, w.z, w.w };
#pragma unroll
            for (int t = 0; t < 4; t++) {
                s += bf2f((unsigned short)(d[t] & 0xFFFFu)) * W_L[q * 8 + t * 2];
                s += bf2f((unsigned short)(d[t] >> 16))     * W_L[q * 8 + t * 2 + 1];
            }
        }
#pragma unroll
        for (int q = 0; q < 8; q++) {
            uint4 w = hc[q];
            uint32_t d[4] = { w.x, w.y, w.z, w.w };
#pragma unroll
            for (int t = 0; t < 4; t++) {
                s += bf2f((unsigned short)(d[t] & 0xFFFFu)) * W_L[64 + q * 8 + t * 2];
                s += bf2f((unsigned short)(d[t] >> 16))     * W_L[64 + q * 8 + t * 2 + 1];
            }
        }
        atomicAddF(&Lz[r], s * (x[c] - mean[c]));
        if (i < N) logv = logf(fabsf(s));
    }
    __shared__ float red[256];
    red[threadIdx.x] = logv;
    __syncthreads();
    for (int off = 128; off > 0; off >>= 1) {
        if (threadIdx.x < off) red[threadIdx.x] += red[threadIdx.x + off];
        __syncthreads();
    }
    if (threadIdx.x == 0) atomicAddF(&accum[0], red[0]);
}

// ---------- xPx = sum(Lz^2) ----------
__global__ void xpx_kernel(const float* __restrict__ Lz, float* __restrict__ accum, int N) {
    int i = blockIdx.x * blockDim.x + threadIdx.x;
    float v = (i < N) ? Lz[i] * Lz[i] : 0.0f;
    __shared__ float red[256];
    red[threadIdx.x] = v;
    __syncthreads();
    for (int off = 128; off > 0; off >>= 1) {
        if (threadIdx.x < off) red[threadIdx.x] += red[threadIdx.x + off];
        __syncthreads();
    }
    if (threadIdx.x == 0) atomicAddF(&accum[1], red[0]);
}

__global__ void final_kernel(const float* __restrict__ accum, float* __restrict__ out) {
    out[0] = accum[0] - 0.5f * accum[1];
}

// ---------- host launcher ----------
extern "C" void kernel_launch(void* const* d_in, const int* in_sizes, int n_in,
                              void* d_out, int out_size, void* d_ws, size_t ws_size,
                              hipStream_t stream) {
    const float* x    = (const float*)d_in[0];
    const float* nf   = (const float*)d_in[1];
    const float* ea   = (const float*)d_in[2];
    const float* mean = (const float*)d_in[3];
    const float* Wemb = (const float*)d_in[4];
    const float* bemb = (const float*)d_in[5];
    const float* We1  = (const float*)d_in[6];
    const float* be1  = (const float*)d_in[7];
    const float* We2  = (const float*)d_in[8];
    const float* be2  = (const float*)d_in[9];
    const float* Wn1  = (const float*)d_in[10];
    const float* bn1  = (const float*)d_in[11];
    const float* Wn2  = (const float*)d_in[12];
    const float* bn2  = (const float*)d_in[13];
    const float* W_L  = (const float*)d_in[14];
    const float* b_L  = (const float*)d_in[15];
    const int* ei     = (const int*)d_in[16];
    const int* Lrow   = (const int*)d_in[17];
    const int* Lcol   = (const int*)d_in[18];

    const int N = in_sizes[0];
    const int E = in_sizes[2] / 4;
    const int M = in_sizes[17];
    const int H = 64;

    char* ws = (char*)d_ws;
    size_t off = 0;
    auto take = [&](size_t bytes) -> void* {
        void* p = ws + off;
        off += (bytes + 255) & ~(size_t)255;
        return p;
    };
    unsigned short* hA = (unsigned short*)take((size_t)N * H * 2);
    unsigned short* hB = (unsigned short*)take((size_t)N * H * 2);
    float* aggr        = (float*)take((size_t)N * H * 4);
    float* Lz          = (float*)take((size_t)N * 4);
    uint32_t* Wef      = (uint32_t*)take(2 * EW_DW * 4);       // [layer][W1(5120)|W2(2048)]
    uint32_t* Wn1f     = (uint32_t*)take(2 * 4096 * 4);
    uint32_t* Wn2f     = (uint32_t*)take(2 * 2048 * 4);
    float* accum       = (float*)take(2 * 4);

    // pack all MLP weights to bf16 WMMA-fragment order
    for (int l = 0; l < 2; l++) {
        pack_frag_kernel<<<(5 * 1024 + 255) / 256, 256, 0, stream>>>(
            We1 + l * 132 * 64, Wef + l * EW_DW, 132, 5);
        pack_frag_kernel<<<(2 * 1024 + 255) / 256, 256, 0, stream>>>(
            We2 + l * 64 * 64, Wef + l * EW_DW + 5 * 1024, 64, 2);
        pack_frag_kernel<<<(4 * 1024 + 255) / 256, 256, 0, stream>>>(
            Wn1 + l * 128 * 64, Wn1f + l * 4096, 128, 4);
        pack_frag_kernel<<<(2 * 1024 + 255) / 256, 256, 0, stream>>>(
            Wn2 + l * 64 * 64, Wn2f + l * 2048, 64, 2);
    }

    embed_kernel<<<(N * H + 255) / 256, 256, 0, stream>>>(nf, Wemb, bemb, hA, N);

    const int* esrc = ei;
    const int* edst = ei + E;
    unsigned short* hin = hA;
    unsigned short* hout = hB;
    for (int l = 0; l < 2; l++) {
        (void)hipMemsetAsync(aggr, 0, (size_t)N * H * 4, stream);
        int etiles = (E + 15) / 16, eblocks = (etiles + WAVES - 1) / WAVES;
        edge_mlp_kernel<<<eblocks, 128, 0, stream>>>(hin, ea, esrc, edst,
            Wef + l * EW_DW, be1 + l * 64, be2 + l * 64, aggr, N, E);
        int ntiles = (N + 15) / 16, nblocks = (ntiles + WAVES - 1) / WAVES;
        node_mlp_kernel<<<nblocks, 128, 0, stream>>>(hin, aggr,
            Wn1f + l * 4096, bn1 + l * 64, Wn2f + l * 2048, bn2 + l * 64, hout, N);
        unsigned short* t = hin; hin = hout; hout = t;
    }

    (void)hipMemsetAsync(Lz, 0, (size_t)N * 4, stream);
    (void)hipMemsetAsync(accum, 0, 2 * 4, stream);
    factor_kernel<<<(M + 255) / 256, 256, 0, stream>>>(hin, W_L, b_L, Lrow, Lcol,
                                                       x, mean, Lz, accum, N, M);
    xpx_kernel<<<(N + 255) / 256, 256, 0, stream>>>(Lz, accum, N);
    final_kernel<<<1, 1, 0, stream>>>(accum, (float*)d_out);
}
